// VSSBlock_8177617731786
// MI455X (gfx1250) — matrix operand order repaired and checked by reading, not compiled, not run on hardware
//
#include <hip/hip_runtime.h>
#include <hip/hip_bf16.h>
#include <math.h>

#define B_    8
#define H_    56
#define W_    56
#define C_    96
#define D_    192
#define NS    16
#define KD    4
#define R_    6
#define L_    (H_ * W_)          // 3136
#define MROWS (B_ * L_)          // 25088
#define MLPH  384
#define EPS_  1e-5f
#define PROJN 192                // x_proj N padded: per-k 48 cols: dt@0..5, B@8..23, C@24..39
#define KMAX  384                // largest GEMM K (for LDS staging buffer)

#define AS1 __attribute__((address_space(1)))
#define AS3 __attribute__((address_space(3)))

typedef __attribute__((ext_vector_type(16))) _Float16 v16h;
typedef __attribute__((ext_vector_type(8)))  float    v8f;
typedef __attribute__((ext_vector_type(4)))  float    v4f;
typedef __attribute__((ext_vector_type(4)))  int      v4i;

static __device__ __forceinline__ float sigmoidf_(float x) {
    return 1.0f / (1.0f + __expf(-x));
}

// ---------------------------------------------------------------------------
// WMMA GEMM:  C[M,N] = A[M,K] (f16 row-major) x W[N,K]^T (f16 row-major)
// 4-wave block; weight strip W[n0:n0+16, :] staged into LDS once via
// global_load_async_to_lds_b128; each wave computes a 64x16 C tile.
// Requires M%256==0, N%16==0, K%32==0, K<=KMAX.
// ---------------------------------------------------------------------------
__global__ __launch_bounds__(128)
void wmma_gemm(const _Float16* __restrict__ A, const _Float16* __restrict__ W,
               float* __restrict__ C, int M, int N, int K) {
    __shared__ __align__(32) _Float16 wsh[16 * KMAX];

    const int n0   = blockIdx.y * 16;
    const int wave = threadIdx.x >> 5;
    const int lane = threadIdx.x & 31;
    const int g    = lane >> 4;   // lane group 0/1
    const int col  = lane & 15;
    const int m0   = blockIdx.x * 256 + wave * 64;

    // ---- async-stage the 16 x K weight strip into LDS (contiguous rows) ----
    {
        const int totalB = 16 * K * 2;
        const char* gsrc = (const char*)(W + (size_t)n0 * K);
        for (int off = threadIdx.x * 16; off < totalB; off += 128 * 16) {
            __builtin_amdgcn_global_load_async_to_lds_b128(
                (AS1 v4i*)(gsrc + off),
                (AS3 v4i*)((char*)wsh + off), 0, 0);
        }
        __builtin_amdgcn_s_wait_asynccnt(0);
        __syncthreads();
    }

    v8f acc0 = {}, acc1 = {}, acc2 = {}, acc3 = {};

    for (int k0 = 0; k0 < K; k0 += 32) {
        // B fragment from LDS: lane holds column (n0+col), K-run g*16..g*16+15
        v16h bf = *(const v16h*)&wsh[col * K + k0 + g * 16];

        v16h af;
        const _Float16* ap;
#define DO_TILE(ACC, I)                                                        \
        ap = A + (size_t)(m0 + (I)*16 + col) * K + k0;                         \
        ((v4f*)&af)[0] = *(const v4f*)(ap + g * 16);                           \
        ((v4f*)&af)[1] = *(const v4f*)(ap + g * 16 + 8);                       \
        ACC = __builtin_amdgcn_wmma_f32_16x16x32_f16(false, af, false, bf,     \
                                                     (short)0, ACC, false, false);
        DO_TILE(acc0, 0)
        DO_TILE(acc1, 1)
        DO_TILE(acc2, 2)
        DO_TILE(acc3, 3)
#undef DO_TILE
    }

#define ST_TILE(ACC, I)                                                        \
    {                                                                          \
        int mb = m0 + (I)*16 + g * 8;                                          \
        _Pragma("unroll")                                                      \
        for (int r = 0; r < 8; ++r)                                            \
            C[(size_t)(mb + r) * N + n0 + col] = ACC[r];                       \
    }
    ST_TILE(acc0, 0)
    ST_TILE(acc1, 1)
    ST_TILE(acc2, 2)
    ST_TILE(acc3, 3)
#undef ST_TILE
}

// ---------------------------------------------------------------------------
// LayerNorm (one wave32 per row) -> f16
// ---------------------------------------------------------------------------
template <int CC>
__global__ void ln_to_f16(const float* __restrict__ x, const float* __restrict__ w,
                          const float* __restrict__ b, _Float16* __restrict__ out,
                          int rows) {
    int gid  = blockIdx.x * blockDim.x + threadIdx.x;
    int wave = gid >> 5;
    int lane = gid & 31;
    if (wave >= rows) return;
    constexpr int PER = CC / 32;
    const float* xp = x + (size_t)wave * CC;

    float v[PER];
    float s = 0.f;
#pragma unroll
    for (int i = 0; i < PER; ++i) { v[i] = xp[lane + 32 * i]; s += v[i]; }
#pragma unroll
    for (int m = 16; m; m >>= 1) s += __shfl_xor(s, m, 32);
    float mean = s / CC;
    float s2 = 0.f;
#pragma unroll
    for (int i = 0; i < PER; ++i) { float d = v[i] - mean; s2 += d * d; }
#pragma unroll
    for (int m = 16; m; m >>= 1) s2 += __shfl_xor(s2, m, 32);
    float inv = rsqrtf(s2 / CC + EPS_);
#pragma unroll
    for (int i = 0; i < PER; ++i) {
        int c = lane + 32 * i;
        out[(size_t)wave * CC + c] = (_Float16)((v[i] - mean) * inv * w[c] + b[c]);
    }
}

// ---------------------------------------------------------------------------
// Depthwise 3x3 conv + bias + SiLU. In: xz (row, 2D) channels [0,D). Out f16.
// ---------------------------------------------------------------------------
__global__ void conv_silu(const float* __restrict__ xz, const float* __restrict__ cw,
                          const float* __restrict__ cb, _Float16* __restrict__ out) {
    int idx = blockIdx.x * blockDim.x + threadIdx.x;
    if (idx >= MROWS * D_) return;
    int d  = idx % D_;
    int bl = idx / D_;
    int l  = bl % L_;
    int b  = bl / L_;
    int h  = l / W_, w = l % W_;
    float acc = cb[d];
#pragma unroll
    for (int dh = -1; dh <= 1; ++dh) {
#pragma unroll
        for (int dw = -1; dw <= 1; ++dw) {
            int hh = h + dh, ww = w + dw;
            if (hh >= 0 && hh < H_ && ww >= 0 && ww < W_)
                acc += xz[((size_t)b * L_ + hh * W_ + ww) * (2 * D_) + d] *
                       cw[d * 9 + (dh + 1) * 3 + (dw + 1)];
        }
    }
    out[idx] = (_Float16)(acc * sigmoidf_(acc));
}

// ---------------------------------------------------------------------------
// dt = softplus(dt_proj_w[k,d,:] . proj[row,k,0:6] + dt_proj_b[k,d])
// ---------------------------------------------------------------------------
__global__ void dt_kernel(const float* __restrict__ proj, const float* __restrict__ dtw,
                          const float* __restrict__ dtb, float* __restrict__ dtbuf) {
    int idx = blockIdx.x * blockDim.x + threadIdx.x;
    if (idx >= MROWS * KD * D_) return;
    int d   = idx % D_;
    int k   = (idx / D_) % KD;
    size_t row = idx / (D_ * KD);
    const float* pr = proj + (row * KD + k) * 48;
    float acc = dtb[k * D_ + d];
#pragma unroll
    for (int r = 0; r < R_; ++r) acc += dtw[((k * D_ + d) * R_) + r] * pr[r];
    float sp = (acc > 20.f) ? acc : log1pf(__expf(acc));
    dtbuf[(row * KD + k) * D_ + d] = sp;
}

// ---------------------------------------------------------------------------
// Selective scan, one thread per (b,k,d) chain; h[16] in registers.
// Software-pipelined: step l+1 loads are issued before step l's exp chain.
// B/C read as aligned v4f (proj per-k layout: dt@0..5, B@8..23, C@24..39).
// Fused cross-merge via global_atomic_add_f32 at permuted position p.
// ---------------------------------------------------------------------------
struct ScanStep {
    float xv, dt;
    v4f bv[4], cv[4];
    size_t row;
};

__global__ void ss_scan(const _Float16* __restrict__ xconv, const float* __restrict__ dtbuf,
                        const float* __restrict__ proj, const float* __restrict__ A_logs,
                        const float* __restrict__ Ds, float* __restrict__ ym) {
    int tid = blockIdx.x * blockDim.x + threadIdx.x;
    if (tid >= B_ * KD * D_) return;
    int d = tid % D_;
    int k = (tid / D_) % KD;
    int b = tid / (D_ * KD);

    float a[NS], h[NS];
#pragma unroll
    for (int n = 0; n < NS; ++n) {
        a[n] = -__expf(A_logs[(size_t)(k * D_ + d) * NS + n]);
        h[n] = 0.f;
    }
    float dsv = Ds[k * D_ + d];

    auto loadstep = [&](int l, ScanStep& s) {
        int l2 = (k & 2) ? (L_ - 1 - l) : l;                 // k=2,3: reversed
        int p  = (k & 1) ? ((l2 % H_) * W_ + l2 / H_) : l2;  // k=1,3: transposed
        size_t row = (size_t)b * L_ + p;
        s.row = row;
        s.xv  = (float)xconv[row * D_ + d];
        s.dt  = dtbuf[(row * KD + k) * D_ + d];
        const v4f* bc = (const v4f*)(proj + (row * KD + k) * 48);
#pragma unroll
        for (int i = 0; i < 4; ++i) { s.bv[i] = bc[2 + i]; s.cv[i] = bc[6 + i]; }
    };

    ScanStep cur, nxt;
    loadstep(0, cur);
    for (int l = 0; l < L_; ++l) {
        if (l + 1 < L_) loadstep(l + 1, nxt);   // prefetch next step
        float dx = cur.dt * cur.xv;
        float y = 0.f;
        const float* bp = (const float*)cur.bv;
        const float* cp = (const float*)cur.cv;
#pragma unroll
        for (int n = 0; n < NS; ++n) {
            h[n] = h[n] * __expf(cur.dt * a[n]) + dx * bp[n];
            y += h[n] * cp[n];
        }
        atomicAdd(&ym[cur.row * D_ + d], y + dsv * cur.xv);
        cur = nxt;
    }
}

// ---------------------------------------------------------------------------
// out-norm (LN over D) + gate with silu(z); one wave per row -> f16
// ---------------------------------------------------------------------------
__global__ void outnorm_gate(const float* __restrict__ ym, const float* __restrict__ w,
                             const float* __restrict__ b, const float* __restrict__ xz,
                             _Float16* __restrict__ out, int rows) {
    int gid  = blockIdx.x * blockDim.x + threadIdx.x;
    int wave = gid >> 5;
    int lane = gid & 31;
    if (wave >= rows) return;
    const float* yp = ym + (size_t)wave * D_;
    float v[6];
    float s = 0.f;
#pragma unroll
    for (int i = 0; i < 6; ++i) { v[i] = yp[lane + 32 * i]; s += v[i]; }
#pragma unroll
    for (int m = 16; m; m >>= 1) s += __shfl_xor(s, m, 32);
    float mean = s / D_;
    float s2 = 0.f;
#pragma unroll
    for (int i = 0; i < 6; ++i) { float d = v[i] - mean; s2 += d * d; }
#pragma unroll
    for (int m = 16; m; m >>= 1) s2 += __shfl_xor(s2, m, 32);
    float inv = rsqrtf(s2 / D_ + EPS_);
#pragma unroll
    for (int i = 0; i < 6; ++i) {
        int c = lane + 32 * i;
        float val = (v[i] - mean) * inv * w[c] + b[c];
        float z = xz[(size_t)wave * (2 * D_) + D_ + c];
        out[(size_t)wave * D_ + c] = (_Float16)(val * z * sigmoidf_(z));
    }
}

// ---------------------------------------------------------------------------
// small elementwise helpers
// ---------------------------------------------------------------------------
__global__ void f32_to_f16(const float* __restrict__ s, _Float16* __restrict__ d, int n) {
    int i = blockIdx.x * blockDim.x + threadIdx.x;
    if (i < n) d[i] = (_Float16)s[i];
}

// re-pack x_proj_w (4,38,192) into (192,192) f16 with per-k 48-row groups:
// dt rows @ c 0..5, B rows @ c 8..23, C rows @ c 24..39, rest zero.
__global__ void pad_xproj(const float* __restrict__ s, _Float16* __restrict__ d) {
    int idx = blockIdx.x * blockDim.x + threadIdx.x;
    if (idx >= PROJN * D_) return;
    int kk = idx % D_;
    int rr = idx / D_;
    int k = rr / 48, c = rr % 48;
    int sc = -1;
    if (c < R_)                 sc = c;                       // dt rows
    else if (c >= 8 && c < 24)  sc = R_ + (c - 8);            // B rows
    else if (c >= 24 && c < 40) sc = R_ + NS + (c - 24);      // C rows
    float v = (sc >= 0) ? s[((size_t)k * (R_ + 2 * NS) + sc) * D_ + kk] : 0.f;
    d[idx] = (_Float16)v;
}

__global__ void zero_f32(float* __restrict__ p, int n) {
    int i = blockIdx.x * blockDim.x + threadIdx.x;
    if (i < n) p[i] = 0.f;
}

__global__ void resid_add(const float* __restrict__ a, const float* __restrict__ b,
                          float* __restrict__ o, int n) {
    int i = blockIdx.x * blockDim.x + threadIdx.x;
    if (i < n) o[i] = a[i] + b[i];
}

__global__ void bias_gelu_f16(const float* __restrict__ a, const float* __restrict__ bias,
                              _Float16* __restrict__ o, int n, int cols) {
    int i = blockIdx.x * blockDim.x + threadIdx.x;
    if (i >= n) return;
    float x = a[i] + bias[i % cols];
    // 0.5*(1+tanh(u)) == sigmoid(2u)
    float u2 = 1.5957691216f * (x + 0.044715f * x * x * x);
    o[i] = (_Float16)(x * sigmoidf_(u2));
}

__global__ void final_add(const float* __restrict__ x1, const float* __restrict__ a2,
                          const float* __restrict__ bias, float* __restrict__ o,
                          int n, int cols) {
    int i = blockIdx.x * blockDim.x + threadIdx.x;
    if (i < n) o[i] = x1[i] + a2[i] + bias[i % cols];
}

// ---------------------------------------------------------------------------
static inline int cdiv(int a, int b) { return (a + b - 1) / b; }

extern "C" void kernel_launch(void* const* d_in, const int* in_sizes, int n_in,
                              void* d_out, int out_size, void* d_ws, size_t ws_size,
                              hipStream_t stream) {
    const float* x         = (const float*)d_in[0];
    const float* ln1_w     = (const float*)d_in[1];
    const float* ln1_b     = (const float*)d_in[2];
    const float* in_proj_w = (const float*)d_in[3];
    const float* conv_w    = (const float*)d_in[4];
    const float* conv_b    = (const float*)d_in[5];
    const float* x_proj_w  = (const float*)d_in[6];
    const float* dt_proj_w = (const float*)d_in[7];
    const float* dt_proj_b = (const float*)d_in[8];
    const float* A_logs    = (const float*)d_in[9];
    const float* Ds        = (const float*)d_in[10];
    const float* onorm_w   = (const float*)d_in[11];
    const float* onorm_b   = (const float*)d_in[12];
    const float* out_proj_w= (const float*)d_in[13];
    const float* ln2_w     = (const float*)d_in[14];
    const float* ln2_b     = (const float*)d_in[15];
    const float* fc1_w     = (const float*)d_in[16];
    const float* fc1_b     = (const float*)d_in[17];
    const float* fc2_w     = (const float*)d_in[18];
    const float* fc2_b     = (const float*)d_in[19];
    float* out = (float*)d_out;

    uint8_t* base = (uint8_t*)d_ws;
    size_t off = 0;
    auto alloc = [&](size_t bytes) -> void* {
        void* p = base + off;
        off += (bytes + 255) & ~(size_t)255;
        return p;
    };

    _Float16* t16      = (_Float16*)alloc((size_t)MROWS * C_ * 2);
    _Float16* w_in16   = (_Float16*)alloc((size_t)2 * D_ * C_ * 2);
    float*    xz       = (float*)   alloc((size_t)MROWS * 2 * D_ * 4);
    _Float16* xconv16  = (_Float16*)alloc((size_t)MROWS * D_ * 2);
    _Float16* w_proj16 = (_Float16*)alloc((size_t)PROJN * D_ * 2);
    float*    proj     = (float*)   alloc((size_t)MROWS * PROJN * 4);
    float*    dtbuf    = (float*)   alloc((size_t)MROWS * KD * D_ * 4);
    float*    ym       = (float*)   alloc((size_t)MROWS * D_ * 4);
    _Float16* yo16     = (_Float16*)alloc((size_t)MROWS * D_ * 2);
    _Float16* w_out16  = (_Float16*)alloc((size_t)C_ * D_ * 2);
    float*    tmp1     = (float*)   alloc((size_t)MROWS * C_ * 4);
    float*    x1       = (float*)   alloc((size_t)MROWS * C_ * 4);
    _Float16* h16      = (_Float16*)alloc((size_t)MROWS * C_ * 2);
    _Float16* w_fc1_16 = (_Float16*)alloc((size_t)MLPH * C_ * 2);
    float*    a1       = (float*)   alloc((size_t)MROWS * MLPH * 4);
    _Float16* g16      = (_Float16*)alloc((size_t)MROWS * MLPH * 2);
    _Float16* w_fc2_16 = (_Float16*)alloc((size_t)C_ * MLPH * 2);
    float*    a2       = (float*)   alloc((size_t)MROWS * C_ * 4);

    const int TB = 256;

    // ---- weight conversions (f32 -> f16) ----
    f32_to_f16<<<cdiv(2 * D_ * C_, TB), TB, 0, stream>>>(in_proj_w, w_in16, 2 * D_ * C_);
    pad_xproj <<<cdiv(PROJN * D_, TB), TB, 0, stream>>>(x_proj_w, w_proj16);
    f32_to_f16<<<cdiv(C_ * D_, TB), TB, 0, stream>>>(out_proj_w, w_out16, C_ * D_);
    f32_to_f16<<<cdiv(MLPH * C_, TB), TB, 0, stream>>>(fc1_w, w_fc1_16, MLPH * C_);
    f32_to_f16<<<cdiv(C_ * MLPH, TB), TB, 0, stream>>>(fc2_w, w_fc2_16, C_ * MLPH);

    // ---- LN1 -> f16 ----
    ln_to_f16<C_><<<cdiv(MROWS * 32, TB), TB, 0, stream>>>(x, ln1_w, ln1_b, t16, MROWS);

    // ---- in_proj GEMM: (25088 x 384) ----
    wmma_gemm<<<dim3(MROWS / 256, (2 * D_) / 16), 128, 0, stream>>>(t16, w_in16, xz,
                                                                    MROWS, 2 * D_, C_);

    // ---- depthwise conv + SiLU -> f16 ----
    conv_silu<<<cdiv(MROWS * D_, TB), TB, 0, stream>>>(xz, conv_w, conv_b, xconv16);

    // ---- x_proj GEMM: (25088 x 192pad) ----
    wmma_gemm<<<dim3(MROWS / 256, PROJN / 16), 128, 0, stream>>>(xconv16, w_proj16, proj,
                                                                 MROWS, PROJN, D_);

    // ---- dt = softplus(...) ----
    dt_kernel<<<cdiv(MROWS * KD * D_, TB), TB, 0, stream>>>(proj, dt_proj_w, dt_proj_b, dtbuf);

    // ---- selective scan with fused cross-merge ----
    zero_f32<<<cdiv(MROWS * D_, TB), TB, 0, stream>>>(ym, MROWS * D_);
    ss_scan<<<cdiv(B_ * KD * D_, TB), TB, 0, stream>>>(xconv16, dtbuf, proj, A_logs, Ds, ym);

    // ---- out-norm + silu(z) gate -> f16 ----
    outnorm_gate<<<cdiv(MROWS * 32, TB), TB, 0, stream>>>(ym, onorm_w, onorm_b, xz, yo16, MROWS);

    // ---- out_proj GEMM + residual ----
    wmma_gemm<<<dim3(MROWS / 256, C_ / 16), 128, 0, stream>>>(yo16, w_out16, tmp1,
                                                              MROWS, C_, D_);
    resid_add<<<cdiv(MROWS * C_, TB), TB, 0, stream>>>(x, tmp1, x1, MROWS * C_);

    // ---- LN2 -> MLP ----
    ln_to_f16<C_><<<cdiv(MROWS * 32, TB), TB, 0, stream>>>(x1, ln2_w, ln2_b, h16, MROWS);
    wmma_gemm<<<dim3(MROWS / 256, MLPH / 16), 128, 0, stream>>>(h16, w_fc1_16, a1,
                                                                MROWS, MLPH, C_);
    bias_gelu_f16<<<cdiv(MROWS * MLPH, TB), TB, 0, stream>>>(a1, fc1_b, g16,
                                                             MROWS * MLPH, MLPH);
    wmma_gemm<<<dim3(MROWS / 256, C_ / 16), 128, 0, stream>>>(g16, w_fc2_16, a2,
                                                              MROWS, C_, MLPH);
    final_add<<<cdiv(MROWS * C_, TB), TB, 0, stream>>>(x1, a2, fc2_b, out,
                                                       MROWS * C_, C_);
}
